// MambaLayer1_13030930776785
// MI455X (gfx1250) — matrix operand order repaired and checked by reading, not compiled, not run on hardware
//
#include <hip/hip_runtime.h>
#include <stdint.h>
#include <stddef.h>

// Problem dims (fixed by reference)
#define B_    4
#define C_    96
#define L_    4096          // H*W
#define HW_   64
#define DI_   192           // D_INNER
#define DS_   16            // D_STATE
#define DTR_  6             // DT_RANK
#define NXZ_  384           // 2*D_INNER
#define NDBL_ 38            // DT_RANK + 2*D_STATE (valid cols)
#define XDBS_ 48            // padded x_dbl row stride
#define KCV_  864           // C*3*3
#define M_    16384         // B*L
#define BN_EPS 1e-5f

typedef __attribute__((ext_vector_type(16))) __bf16    v16bf;
typedef __attribute__((ext_vector_type(8)))  float     v8f;
typedef __attribute__((ext_vector_type(4)))  uint32_t  v4u;
typedef __attribute__((ext_vector_type(8)))  int       v8i;
typedef __attribute__((ext_vector_type(4)))  int       v4i;
typedef __attribute__((ext_vector_type(4)))  float     f4;

union FragBF { v16bf v; uint32_t u[8]; };

__device__ __forceinline__ uint16_t f2bf(float f) {
  union { float f; uint32_t u; } a; a.f = f;
  uint32_t u = a.u;
  u += 0x7FFFu + ((u >> 16) & 1u);   // round-to-nearest-even
  return (uint16_t)(u >> 16);
}

__device__ __forceinline__ float silu_(float x) { return x / (1.0f + __expf(-x)); }

// ---------------------------------------------------------------------------
// Tensor Data Mover: stage a 2D tile (rows x 64B) into LDS.
// D# packing per CDNA5 ISA.
// ---------------------------------------------------------------------------
__device__ __forceinline__ void tdm_load_2d_to_lds(uint32_t lds_off, const void* g,
                                                   uint32_t tile_d0, uint32_t tile_d1,
                                                   uint32_t tens_d0, uint32_t tens_d1,
                                                   uint32_t stride0) {
  uint64_t ga = (uint64_t)(uintptr_t)g;
  v4u g0;
  g0[0] = 1u;                                            // count=1, user mode
  g0[1] = lds_off;                                       // lds_addr (bytes)
  g0[2] = (uint32_t)ga;                                  // global_addr[31:0]
  g0[3] = ((uint32_t)(ga >> 32) & 0x01ffffffu) | 0x80000000u; // addr[56:32] | type=2
  v8i g1;
  g1[0] = (int)(3u << 16);                               // data_size = 8 bytes
  g1[1] = (int)((tens_d0 & 0xffffu) << 16);              // tensor_dim0[15:0]
  g1[2] = (int)((tens_d0 >> 16) | ((tens_d1 & 0xffffu) << 16));
  g1[3] = (int)((tens_d1 >> 16) | (tile_d0 << 16));      // tile_dim0
  g1[4] = (int)(tile_d1 & 0xffffu);                      // tile_dim1 (tile_dim2=0)
  g1[5] = (int)stride0;                                  // tensor_dim0_stride lo32
  g1[6] = 0;
  g1[7] = 0;
  v4i z = {0, 0, 0, 0};
#if __clang_major__ >= 23
  v8i z8 = {0, 0, 0, 0, 0, 0, 0, 0};
  __builtin_amdgcn_tensor_load_to_lds(g0, g1, z, z, z8, 0);
#else
  __builtin_amdgcn_tensor_load_to_lds(g0, g1, z, z, 0);
#endif
}

// ---------------------------------------------------------------------------
// Strip GEMM with global-memory B (x_proj only; weights pre-padded so the
// n-range is always fully valid -> no divergent guards).
// ---------------------------------------------------------------------------
template <int NT>
__device__ __forceinline__ void strip_gemm(const uint16_t* __restrict__ A, int lda,
                                           const uint16_t* __restrict__ W, int ldw,
                                           int m0, int n0, int K, v8f* acc) {
  const int lane = threadIdx.x & 31;
  const int lo = lane & 15, half = lane >> 4;
  const size_t arow = (size_t)(m0 + lo) * (size_t)lda;
  for (int kk = 0; kk < K; kk += 32) {
    FragBF a;
    if (kk + 32 < K) __builtin_prefetch(A + arow + kk + 32, 0, 1);
#pragma unroll
    for (int v = 0; v < 8; ++v) {
      const int ka = kk + (half << 4) + (v << 1);
      a.u[v] = *(const uint32_t*)(A + arow + ka);
    }
#pragma unroll
    for (int t = 0; t < NT; ++t) {
      const int nr = n0 + t * 16 + lo;
      FragBF b;
#pragma unroll
      for (int v = 0; v < 8; ++v) {
        const int kb = kk + (half << 4) + (v << 1);
        b.u[v] = *(const uint32_t*)(W + (size_t)nr * ldw + kb);
      }
      acc[t] = __builtin_amdgcn_wmma_f32_16x16x32_bf16(false, a.v, false, b.v,
                                                       (short)0, acc[t], false, false);
    }
  }
}

// ---------------------------------------------------------------------------
// Panel GEMM: block = 8 waves x 16 rows = 128 rows, N = 96 (6 tiles/wave).
// Weight panel [96 x 32] bf16 (6KB) TDM-staged to LDS, double-buffered.
// EPI: 0 = row-major store (ld=ldout, col offset from blockIdx.y)
//      1 = NCHW store ; 2 = NCHW + BN + ReLU6
// For EPI>=1 the C tile is transposed through wave-private LDS so each lane
// stores 16 consecutive pixels as 4x b128 (coalesced NCHW writes).
// ---------------------------------------------------------------------------
template <int KTOT, int EPI>
__global__ void k_gemm_panel(const uint16_t* __restrict__ A,
                             const uint16_t* __restrict__ W,
                             const float* __restrict__ scale,
                             const float* __restrict__ shift,
                             float* __restrict__ out, int ldout) {
  __shared__ uint16_t panel[2][C_ * 32];                 // 2 x 6 KB
  const int wavein = threadIdx.x >> 5;
  const int lane = threadIdx.x & 31;
  const int lo = lane & 15, half = lane >> 4;
  const int m0 = (blockIdx.x * 8 + wavein) * 16;
  const int n0 = blockIdx.y * C_;
  W += (size_t)blockIdx.y * C_ * KTOT;                   // N-group's weight rows
  const size_t arow = (size_t)(m0 + lo) * KTOT;
  v8f acc[6] = {};
  const int NSLAB = KTOT / 32;

  // make 'panel' compiler-visibly written (TDM writes are invisible to AA)
  for (int i = threadIdx.x; i < C_ * 32; i += 256) {
    panel[0][i] = 0;
    panel[1][i] = 0;
  }
  __syncthreads();
  if (wavein == 0)
    tdm_load_2d_to_lds(0u, W, 8, C_, KTOT / 4, C_, KTOT / 4);

  for (int slab = 0; slab < NSLAB; ++slab) {
    if (wavein == 0) __builtin_amdgcn_s_wait_tensorcnt(0);
    __syncthreads();                                     // panel[slab&1] ready
    asm volatile("" ::: "memory");                       // TDM wrote LDS: reload
    if (wavein == 0 && slab + 1 < NSLAB)
      tdm_load_2d_to_lds((uint32_t)(((slab + 1) & 1) * (C_ * 32 * 2)),
                         W + (size_t)(slab + 1) * 32, 8, C_, KTOT / 4, C_, KTOT / 4);
    const uint16_t* bp = panel[slab & 1];
    const int kk = slab * 32;
    FragBF a;
    if (slab + 1 < NSLAB) __builtin_prefetch(A + arow + kk + 32, 0, 1);
#pragma unroll
    for (int v = 0; v < 8; ++v) {
      const int ka = kk + (half << 4) + (v << 1);
      a.u[v] = *(const uint32_t*)(A + arow + ka);
    }
#pragma unroll
    for (int t = 0; t < 6; ++t) {
      FragBF b;
#pragma unroll
      for (int v = 0; v < 8; ++v)
        b.u[v] = *(const uint32_t*)(bp + (t * 16 + lo) * 32 + (half << 4) + (v << 1));
      acc[t] = __builtin_amdgcn_wmma_f32_16x16x32_bf16(false, a.v, false, b.v,
                                                       (short)0, acc[t], false, false);
    }
  }

  if constexpr (EPI == 0) {
    // row-major: consecutive lanes -> consecutive addresses (already coalesced)
#pragma unroll
    for (int t = 0; t < 6; ++t)
#pragma unroll
      for (int v = 0; v < 8; ++v) {
        const int m = m0 + v + 8 * half;
        out[(size_t)m * ldout + n0 + t * 16 + lo] = acc[t][v];
      }
  } else {
    // NCHW: transpose 16x96 tile through wave-private LDS, then store
    // 16 consecutive pixels per lane as 4x b128.
    __shared__ float tr[8][16][98];                      // 50.2 KB, per-wave slabs
    float (*my)[98] = tr[wavein];
#pragma unroll
    for (int t = 0; t < 6; ++t) {
      const int n = t * 16 + lo;
      float sc = 1.0f, sh = 0.0f;
      if constexpr (EPI == 2) { sc = scale[n]; sh = shift[n]; }
#pragma unroll
      for (int v = 0; v < 8; ++v) {
        float val = acc[t][v];
        if constexpr (EPI == 2) val = fminf(fmaxf(val * sc + sh, 0.0f), 6.0f);
        my[v + 8 * half][n] = val;
      }
    }
    // wave-private data: no barrier needed (DS ops in-order within wave)
    const int bb = m0 >> 12, l0 = m0 & (L_ - 1);
#pragma unroll
    for (int rep = 0; rep < 3; ++rep) {
      const int n = rep * 32 + lane;
      float* dst = out + (((size_t)(bb * C_ + n)) << 12) + l0;
#pragma unroll
      for (int q = 0; q < 4; ++q) {
        f4 val = {my[q * 4 + 0][n], my[q * 4 + 1][n],
                  my[q * 4 + 2][n], my[q * 4 + 3][n]};
        *(f4*)(dst + q * 4) = val;
      }
    }
  }
}

// ---------------------------------------------------------------------------
// Prep kernels
// ---------------------------------------------------------------------------
__global__ void k_cvt_bf16(const float* __restrict__ s, uint16_t* __restrict__ d, int n) {
  int i = blockIdx.x * blockDim.x + threadIdx.x;
  if (i < n) d[i] = f2bf(s[i]);
}

// convert with zero-padding to ntot elements (pads x_proj_w rows 38..47)
__global__ void k_cvt_pad(const float* __restrict__ s, uint16_t* __restrict__ d,
                          int nsrc, int ntot) {
  int i = blockIdx.x * blockDim.x + threadIdx.x;
  if (i < ntot) d[i] = (i < nsrc) ? f2bf(s[i]) : (uint16_t)0;
}

__global__ void k_prep_A(const float* __restrict__ A_log, float* __restrict__ Aneg, int n) {
  int i = blockIdx.x * blockDim.x + threadIdx.x;
  if (i < n) Aneg[i] = -__expf(A_log[i]);
}

__global__ void k_prep_bn(const float* __restrict__ g, const float* __restrict__ b,
                          const float* __restrict__ mean, const float* __restrict__ var,
                          float* __restrict__ scale, float* __restrict__ shift) {
  int i = blockIdx.x * blockDim.x + threadIdx.x;
  if (i < C_) {
    float inv = g[i] * rsqrtf(var[i] + BN_EPS);
    scale[i] = inv;
    shift[i] = b[i] - mean[i] * inv;
  }
}

// x[B,C,H,W] -> seq_bf16[M, C] (token-major)
__global__ void k_pack_seq(const float* __restrict__ x, uint16_t* __restrict__ seq) {
  int i = blockIdx.x * blockDim.x + threadIdx.x;
  if (i >= M_ * C_) return;
  int c = i % C_;
  int m = i / C_;
  int b = m >> 12, l = m & (L_ - 1);
  seq[i] = f2bf(x[((size_t)(b * C_ + c) << 12) + l]);
}

// x_dbl = xi @ x_proj_w^T : padded N=48, padded stride 48 -> no guards
__global__ void k_gemm_xdbl(const uint16_t* __restrict__ A, const uint16_t* __restrict__ W,
                            float* __restrict__ out) {
  int wave = (blockIdx.x * blockDim.x + threadIdx.x) >> 5;   // 0..1023 = tm
  v8f acc[3] = {};
  strip_gemm<3>(A, DI_, W, DI_, wave * 16, 0, DI_, acc);
  int lane = threadIdx.x & 31, lo = lane & 15, half = lane >> 4;
#pragma unroll
  for (int t = 0; t < 3; ++t)
#pragma unroll
    for (int v = 0; v < 8; ++v) {
      int m = wave * 16 + v + 8 * half;
      out[(size_t)m * XDBS_ + t * 16 + lo] = acc[t][v];
    }
}

// ---------------------------------------------------------------------------
// Elementwise / scan kernels
// ---------------------------------------------------------------------------
__global__ void k_conv1d_silu(const float* __restrict__ xz, const float* __restrict__ cw,
                              const float* __restrict__ cb, float* __restrict__ xi,
                              uint16_t* __restrict__ xib) {
  int i = blockIdx.x * blockDim.x + threadIdx.x;
  if (i >= M_ * DI_) return;
  int d = i % DI_;
  int m = i / DI_;
  int l = m & (L_ - 1);
  int mb = m - l;
  float s = cb[d];
#pragma unroll
  for (int j = 0; j < 4; ++j) {
    int lj = l - 3 + j;
    if (lj >= 0) s += cw[d * 4 + j] * xz[(size_t)(mb + lj) * NXZ_ + d];
  }
  float v = silu_(s);
  xi[i] = v;
  xib[i] = f2bf(v);
}

__global__ void k_dt(const float* __restrict__ xdbl, const float* __restrict__ wdt,
                     const float* __restrict__ bdt, float* __restrict__ dt) {
  int i = blockIdx.x * blockDim.x + threadIdx.x;
  if (i >= M_ * DI_) return;
  int d = i % DI_;
  int m = i / DI_;
  float s = bdt[d];
#pragma unroll
  for (int r = 0; r < DTR_; ++r) s += wdt[d * DTR_ + r] * xdbl[(size_t)m * XDBS_ + r];
  dt[i] = (s > 20.0f) ? s : log1pf(__expf(s));
}

// Selective scan: lane g = (b*DI + d)*16 + s ; 16 lanes share (b,d)
__global__ void k_scan(const float* __restrict__ dt, const float* __restrict__ xi,
                       const float* __restrict__ xdbl, const float* __restrict__ Aneg,
                       float* __restrict__ y) {
  int g = blockIdx.x * blockDim.x + threadIdx.x;
  int s  = g & 15;
  int bd = g >> 4;
  int d  = bd % DI_;
  int b  = bd / DI_;
  float Ads = Aneg[d * DS_ + s];
  float h = 0.0f;
  size_t base = (size_t)b * L_;
  for (int t = 0; t < L_; ++t) {
    size_t row = base + t;
    float dtv = dt[row * DI_ + d];
    float xv  = xi[row * DI_ + d];
    float Bv  = xdbl[row * XDBS_ + DTR_ + s];
    float Cv  = xdbl[row * XDBS_ + DTR_ + DS_ + s];
    h = __expf(dtv * Ads) * h + dtv * Bv * xv;
    float contrib = h * Cv;
    contrib += __shfl_xor(contrib, 1);
    contrib += __shfl_xor(contrib, 2);
    contrib += __shfl_xor(contrib, 4);
    contrib += __shfl_xor(contrib, 8);
    if (s == 0) y[row * DI_ + d] = contrib;
  }
}

__global__ void k_gate(const float* __restrict__ y, const float* __restrict__ xi,
                       const float* __restrict__ Dp, const float* __restrict__ xz,
                       uint16_t* __restrict__ yg) {
  int i = blockIdx.x * blockDim.x + threadIdx.x;
  if (i >= M_ * DI_) return;
  int d = i % DI_;
  int m = i / DI_;
  float z = xz[(size_t)m * NXZ_ + DI_ + d];
  yg[i] = f2bf((y[i] + xi[i] * Dp[d]) * silu_(z));
}

__global__ void k_im2col(const float* __restrict__ xm, uint16_t* __restrict__ col) {
  int i = blockIdx.x * blockDim.x + threadIdx.x;
  if (i >= M_ * KCV_) return;
  int k = i % KCV_;
  int m = i / KCV_;
  int b = m >> 12, l = m & (L_ - 1);
  int py = l >> 6, px = l & 63;
  int ic = k / 9, r = k % 9;
  int ky = r / 3, kx = r % 3;
  int yi = py + ky - 1, xj = px + kx - 1;
  uint16_t v = 0;
  if ((unsigned)yi < HW_ && (unsigned)xj < HW_)
    v = f2bf(xm[((size_t)(b * C_ + ic) << 12) + yi * HW_ + xj]);
  col[i] = v;
}

// ---------------------------------------------------------------------------
extern "C" void kernel_launch(void* const* d_in, const int* in_sizes, int n_in,
                              void* d_out, int out_size, void* d_ws, size_t ws_size,
                              hipStream_t stream) {
  (void)in_sizes; (void)n_in; (void)out_size; (void)ws_size;
  const float* x     = (const float*)d_in[0];
  const float* in_w  = (const float*)d_in[1];
  const float* c1w   = (const float*)d_in[2];
  const float* c1b   = (const float*)d_in[3];
  const float* xp_w  = (const float*)d_in[4];
  const float* dt_w  = (const float*)d_in[5];
  const float* dt_b  = (const float*)d_in[6];
  const float* A_log = (const float*)d_in[7];
  const float* Dp    = (const float*)d_in[8];
  const float* out_w = (const float*)d_in[9];
  const float* c2w   = (const float*)d_in[10];
  const float* bng   = (const float*)d_in[11];
  const float* bnb   = (const float*)d_in[12];
  const float* bnm   = (const float*)d_in[13];
  const float* bnv   = (const float*)d_in[14];
  float* out = (float*)d_out;

  char* ws = (char*)d_ws;
  size_t off = 0;
  auto take = [&](size_t bytes) -> void* {
    void* p = ws + off;
    off += (bytes + 255) & ~(size_t)255;
    return p;
  };
  uint16_t* wi   = (uint16_t*)take((size_t)NXZ_ * C_ * 2);
  uint16_t* wxp  = (uint16_t*)take((size_t)XDBS_ * DI_ * 2);  // padded 48 rows
  uint16_t* wo   = (uint16_t*)take((size_t)C_ * DI_ * 2);
  uint16_t* wc   = (uint16_t*)take((size_t)C_ * KCV_ * 2);
  float*    Aneg = (float*)take((size_t)DI_ * DS_ * 4);
  float*    bsc  = (float*)take((size_t)C_ * 4);
  float*    bsh  = (float*)take((size_t)C_ * 4);
  uint16_t* seqb = (uint16_t*)take((size_t)M_ * C_ * 2);
  float*    xz   = (float*)take((size_t)M_ * NXZ_ * 4);
  float*    xi   = (float*)take((size_t)M_ * DI_ * 4);
  uint16_t* xib  = (uint16_t*)take((size_t)M_ * DI_ * 2);
  float*    xdbl = (float*)take((size_t)M_ * XDBS_ * 4);      // padded stride 48
  float*    dt   = (float*)take((size_t)M_ * DI_ * 4);
  float*    ybuf = (float*)take((size_t)M_ * DI_ * 4);
  uint16_t* yg   = (uint16_t*)take((size_t)M_ * DI_ * 2);
  float*    xm   = (float*)take((size_t)M_ * C_ * 4);
  uint16_t* col  = (uint16_t*)take((size_t)M_ * KCV_ * 2);

  const int T = 256;
  auto blocks = [](size_t n, int t) { return (unsigned)((n + t - 1) / t); };

  // weight / constant prep
  k_cvt_bf16<<<blocks((size_t)NXZ_ * C_, T), T, 0, stream>>>(in_w, wi, NXZ_ * C_);
  k_cvt_pad<<<blocks((size_t)XDBS_ * DI_, T), T, 0, stream>>>(xp_w, wxp, NDBL_ * DI_,
                                                              XDBS_ * DI_);
  k_cvt_bf16<<<blocks((size_t)C_ * DI_, T), T, 0, stream>>>(out_w, wo, C_ * DI_);
  k_cvt_bf16<<<blocks((size_t)C_ * KCV_, T), T, 0, stream>>>(c2w, wc, C_ * KCV_);
  k_prep_A<<<blocks((size_t)DI_ * DS_, T), T, 0, stream>>>(A_log, Aneg, DI_ * DS_);
  k_prep_bn<<<1, T, 0, stream>>>(bng, bnb, bnm, bnv, bsc, bsh);

  // mamba block
  k_pack_seq<<<blocks((size_t)M_ * C_, T), T, 0, stream>>>(x, seqb);
  // in_proj: M=16384, N=384 (4 panel groups of 96), K=96
  k_gemm_panel<C_, 0><<<dim3(M_ / 16 / 8, 4), T, 0, stream>>>(seqb, wi, nullptr, nullptr,
                                                              xz, NXZ_);
  k_conv1d_silu<<<blocks((size_t)M_ * DI_, T), T, 0, stream>>>(xz, c1w, c1b, xi, xib);
  k_gemm_xdbl<<<(M_ / 16) * 32 / T, T, 0, stream>>>(xib, wxp, xdbl);
  k_dt<<<blocks((size_t)M_ * DI_, T), T, 0, stream>>>(xdbl, dt_w, dt_b, dt);
  k_scan<<<(B_ * DI_ * DS_) / T, T, 0, stream>>>(dt, xi, xdbl, Aneg, ybuf);
  k_gate<<<blocks((size_t)M_ * DI_, T), T, 0, stream>>>(ybuf, xi, Dp, xz, yg);
  // out_proj: N=96, K=192, NCHW store via LDS transpose
  k_gemm_panel<DI_, 1><<<dim3(M_ / 16 / 8, 1), T, 0, stream>>>(yg, wo, nullptr, nullptr,
                                                               xm, 0);
  // conv3x3 + BN + ReLU6 (implicit GEMM, TDM-staged weights)
  k_im2col<<<blocks((size_t)M_ * KCV_, T), T, 0, stream>>>(xm, col);
  k_gemm_panel<KCV_, 2><<<dim3(M_ / 16 / 8, 1), T, 0, stream>>>(col, wc, bsc, bsh,
                                                                out, 0);
}
